// Model_44135083934114
// MI455X (gfx1250) — matrix operand order repaired and checked by reading, not compiled, not run on hardware
//
#include <hip/hip_runtime.h>

typedef __attribute__((ext_vector_type(16))) __bf16 v16bf;
typedef __attribute__((ext_vector_type(8)))  __bf16 v8bf;
typedef __attribute__((ext_vector_type(8)))  float  v8f;
typedef __attribute__((ext_vector_type(4)))  unsigned int v4u;
typedef __attribute__((ext_vector_type(8)))  unsigned int v8u;
typedef unsigned int u32;

#define SPB   1024
#define NVOX  4096
#define NBLK  16
#define NCO   513
#define NCOP  544   // 17*32, padded freq count

union FragU { v16bf v; v8bf h[2]; };

// A-operand (16x32 bf16): lane holds K = {k0+8h .. k0+8h+7} and {k0+16+8h .. +7}
__device__ __forceinline__ v16bf load_frag_A(const __bf16* rowbase, int k0, int hf) {
  FragU f;
  f.h[0] = *(const v8bf*)(rowbase + k0 + 8 * hf);
  f.h[1] = *(const v8bf*)(rowbase + k0 + 16 + 8 * hf);
  return f.v;
}
// B-operand (32x16 bf16): lane holds K = {k0+8h .. k0+8h+7} and {k0+16+8h .. +7}
__device__ __forceinline__ v16bf load_frag_B(const __bf16* rowbase, int k0, int hf) {
  FragU f;
  f.h[0] = *(const v8bf*)(rowbase + k0 + 8 * hf);
  f.h[1] = *(const v8bf*)(rowbase + k0 + 16 + 8 * hf);
  return f.v;
}

// ---------------------------------------------------------------------------
// Tensor Data Mover: contiguous 1-D global -> LDS DMA (nelem8 x 8-byte elems).
// D# per CDNA5 ISA: group0 {count=1, lds_addr, global_addr,
// type=2}; group1 {data_size=8B, tile_dim0=nelem8, tensor_dim0=stride0=nelem8,
// tensor_dim1=1}. Issued once per workgroup (wave 0); tracked by TENSORcnt.
__device__ __forceinline__ void tdm_load_1d(u32 lds_off, const void* gptr, u32 nelem8) {
  unsigned long long ga = (unsigned long long)(size_t)gptr;
  v4u g0;
  g0.x = 1u;                                        // count=1 (valid user D#)
  g0.y = lds_off;                                   // lds_addr (bytes)
  g0.z = (u32)(ga & 0xFFFFFFFFu);                   // global_addr[31:0]
  g0.w = (u32)((ga >> 32) & 0x01FFFFFFu) | (2u << 30); // global_addr[56:32] | type=2
  v8u g1;
  g1[0] = 3u << 16;                                 // data_size = 8 bytes
  g1[1] = (nelem8 & 0xFFFFu) << 16;                 // tensor_dim0[15:0]
  g1[2] = (nelem8 >> 16) | (1u << 16);              // tensor_dim0[31:16] | tensor_dim1=1
  g1[3] = (nelem8 & 0xFFFFu) << 16;                 // tile_dim0
  g1[4] = 0u;                                       // tile_dim1/2 unused
  g1[5] = nelem8;                                   // tensor_dim0_stride[31:0]
  g1[6] = 0u;
  g1[7] = 0u;
  asm volatile("tensor_load_to_lds %0, %1" :: "s"(g0), "s"(g1) : "memory");
}

// ---------------------------------------------------------------------------
// Fill DFT basis matrices (bf16), ortho norm (1/32), computed every call.
// Forward stored [f][t] (544 x 1024); inverse stored [t][f] (1024 x 544).
__global__ __launch_bounds__(256)
void basis_kernel(__bf16* __restrict__ bfc, __bf16* __restrict__ bfs,
                  __bf16* __restrict__ bic, __bf16* __restrict__ bis) {
  const int o = blockIdx.x * 256 + threadIdx.x;
  if (o >= NCOP * SPB) return;
  const float inv32 = 1.0f / 32.0f;
  const float w0 = 6.135923151e-3f; // 2*pi/1024
  { // forward: row f, col t;  Re basis cos, Im basis -sin
    int f = o >> 10, t = o & 1023;
    int m = (f * t) & 1023;
    float ang = (float)m * w0;
    bfc[o] = (__bf16)( cosf(ang) * inv32);
    bfs[o] = (__bf16)(-sinf(ang) * inv32);
  }
  { // inverse: row t, col f; weight 2 on interior bins, zeros in pad
    int t = o / NCOP, f = o % NCOP;
    float vc = 0.0f, vs = 0.0f;
    if (f < NCO) {
      float wf = (f == 0 || f == 512) ? 1.0f : 2.0f;
      int m = (t * f) & 1023;
      float ang = (float)m * w0;
      vc =  wf * cosf(ang) * inv32;
      vs = -wf * sinf(ang) * inv32;
    }
    bic[o] = (__bf16)vc;
    bis[o] = (__bf16)vs;
  }
}

// ---------------------------------------------------------------------------
// sf += clip(linear_interp(impulses),0,1) * noise ; also emit bf16 mirror
__global__ __launch_bounds__(256)
void add_env_kernel(const float* __restrict__ imp, const float* __restrict__ noise,
                    float* __restrict__ sf, __bf16* __restrict__ sfh,
                    int b, int first) {
  const size_t o = (size_t)blockIdx.x * 256 + threadIdx.x; // < 4096*1024
  const int v = (int)(o >> 10);
  const int t = (int)(o & 1023);
  float c = ((float)t + 0.5f) * (16.0f / 1024.0f) - 0.5f;
  c = fminf(fmaxf(c, 0.0f), 15.0f);
  int lo = (int)c;
  int hi = lo + 1 > 15 ? 15 : lo + 1;
  float w = c - (float)lo;
  const float* ip = imp + ((size_t)b * NVOX + v) * 16;
  float e = ip[lo] * (1.0f - w) + ip[hi] * w;
  e = fminf(fmaxf(e, 0.0f), 1.0f);
  e *= noise[((size_t)b * NVOX + v) * SPB + t];
  float s = (first ? 0.0f : sf[o]) + e;
  sf[o]  = s;
  sfh[o] = (__bf16)s;
}

// ---------------------------------------------------------------------------
// Forward DFT as GEMM: spec_{re,im}(4096x544) = SF(4096x1024) x basis(1024x544)
// fused with clip(transfer_function) multiply. One 16-voxel M-tile per block.
__global__ __launch_bounds__(256)
void fwd_dft_kernel(const __bf16* __restrict__ sfh,
                    const __bf16* __restrict__ bfc,
                    const __bf16* __restrict__ bfs,
                    const float* __restrict__ tf,
                    __bf16* __restrict__ specRe,
                    __bf16* __restrict__ specIm,
                    int b) {
  __shared__ __attribute__((aligned(16))) __bf16 ldsA[16 * SPB]; // 32 KB
  const int tid = threadIdx.x;
  const int m0 = blockIdx.x * 16;
  if (tid < 32) { // wave 0: TDM DMA of contiguous 32KB A tile, global -> LDS
    tdm_load_1d((u32)(size_t)(void*)ldsA, sfh + (size_t)m0 * SPB, (16 * SPB * 2) / 8);
    __builtin_amdgcn_s_wait_tensorcnt(0);
  }
  __syncthreads();
  const int wave = tid >> 5, lane = tid & 31;
  const int hf = lane >> 4, mrow = lane & 15;
  const __bf16* arow = ldsA + (size_t)mrow * SPB;
  for (int nt = wave; nt < NCOP / 16; nt += 8) {
    const int n0 = nt * 16;
    const int fcol = n0 + (lane & 15);
    const __bf16* bcrow = bfc + (size_t)fcol * SPB;
    const __bf16* bsrow = bfs + (size_t)fcol * SPB;
    v8f accRe = {}; v8f accIm = {};
    // software pipeline: B fragments for step k issued one iteration ahead
    v16bf bc = load_frag_B(bcrow, 0, hf);
    v16bf bs = load_frag_B(bsrow, 0, hf);
    for (int k0 = 0; k0 < SPB; k0 += 32) {
      const int kn = (k0 + 32) & (SPB - 1);
      v16bf a    = load_frag_A(arow, k0, hf);
      v16bf bc_n = load_frag_B(bcrow, kn, hf);
      v16bf bs_n = load_frag_B(bsrow, kn, hf);
      accRe = __builtin_amdgcn_wmma_f32_16x16x32_bf16(false, a, false, bc,
                                                      (short)0, accRe, false, false);
      accIm = __builtin_amdgcn_wmma_f32_16x16x32_bf16(false, a, false, bs,
                                                      (short)0, accIm, false, false);
      bc = bc_n; bs = bs_n;
    }
    for (int r = 0; r < 8; ++r) {
      const int m = m0 + hf * 8 + r;
      float rc = 0.0f;
      if (fcol < NCO) {
        float tv = tf[((size_t)b * NVOX + m) * NCO + fcol];
        rc = fminf(fmaxf(tv, 0.0f), 1.0f);
      }
      specRe[(size_t)m * NCOP + fcol] = (__bf16)(accRe[r] * rc);
      specIm[(size_t)m * NCOP + fcol] = (__bf16)(accIm[r] * rc);
    }
  }
}

// ---------------------------------------------------------------------------
// Inverse DFT as GEMM: Y(4096x1024) = Sre(4096x544)xCi^T + Sim x Si^T
__global__ __launch_bounds__(256)
void inv_dft_kernel(const __bf16* __restrict__ specRe,
                    const __bf16* __restrict__ specIm,
                    const __bf16* __restrict__ bic,
                    const __bf16* __restrict__ bis,
                    float* __restrict__ sf_new) {
  __shared__ __attribute__((aligned(16))) __bf16 ldsRe[16 * NCOP];
  __shared__ __attribute__((aligned(16))) __bf16 ldsIm[16 * NCOP];
  const int tid = threadIdx.x;
  const int m0 = blockIdx.x * 16;
  if (tid < 32) { // wave 0: two TDM DMAs (Re/Im spec tiles), then one wait
    tdm_load_1d((u32)(size_t)(void*)ldsRe, specRe + (size_t)m0 * NCOP, (16 * NCOP * 2) / 8);
    tdm_load_1d((u32)(size_t)(void*)ldsIm, specIm + (size_t)m0 * NCOP, (16 * NCOP * 2) / 8);
    __builtin_amdgcn_s_wait_tensorcnt(0);
  }
  __syncthreads();
  const int wave = tid >> 5, lane = tid & 31;
  const int hf = lane >> 4, mrow = lane & 15;
  const __bf16* arRow = ldsRe + (size_t)mrow * NCOP;
  const __bf16* aiRow = ldsIm + (size_t)mrow * NCOP;
  for (int nt = wave; nt < SPB / 16; nt += 8) {
    const int n0 = nt * 16;
    const int tcol = n0 + (lane & 15);
    const __bf16* bcRow = bic + (size_t)tcol * NCOP;
    const __bf16* bsRow = bis + (size_t)tcol * NCOP;
    v8f acc = {};
    v16bf bc = load_frag_B(bcRow, 0, hf);
    v16bf bs = load_frag_B(bsRow, 0, hf);
    for (int k0 = 0; k0 < NCOP; k0 += 32) {
      const int kn = (k0 + 32 < NCOP) ? k0 + 32 : 0;
      v16bf ar   = load_frag_A(arRow, k0, hf);
      v16bf ai   = load_frag_A(aiRow, k0, hf);
      v16bf bc_n = load_frag_B(bcRow, kn, hf);
      v16bf bs_n = load_frag_B(bsRow, kn, hf);
      acc = __builtin_amdgcn_wmma_f32_16x16x32_bf16(false, ar, false, bc,
                                                    (short)0, acc, false, false);
      acc = __builtin_amdgcn_wmma_f32_16x16x32_bf16(false, ai, false, bs,
                                                    (short)0, acc, false, false);
      bc = bc_n; bs = bs_n;
    }
    for (int r = 0; r < 8; ++r) {
      const int m = m0 + hf * 8 + r;
      sf_new[(size_t)m * SPB + tcol] = acc[r];
    }
  }
}

// ---------------------------------------------------------------------------
// mic tap (pre-box) + 3x3x3 zero-padded box sum, sf_new -> sf
__global__ __launch_bounds__(256)
void box_mic_kernel(const float* __restrict__ sf_new, float* __restrict__ sf,
                    float* __restrict__ out, int b) {
  const size_t o = (size_t)blockIdx.x * 256 + threadIdx.x;
  const int v = (int)(o >> 10);
  const int t = (int)(o & 1023);
  const int x = v >> 8, y = (v >> 4) & 15, z = v & 15;
  float s = 0.0f;
  for (int dx = -1; dx <= 1; ++dx) {
    int nx = x + dx; if (nx < 0 || nx > 15) continue;
    for (int dy = -1; dy <= 1; ++dy) {
      int ny = y + dy; if (ny < 0 || ny > 15) continue;
      for (int dz = -1; dz <= 1; ++dz) {
        int nz = z + dz; if (nz < 0 || nz > 15) continue;
        s += sf_new[(((size_t)((nx << 8) | (ny << 4) | nz)) << 10) + t];
      }
    }
  }
  if (v == ((8 << 8) | (8 << 4) | 8)) out[(size_t)b * SPB + t] = sf_new[o];
  sf[o] = s;
}

// ---------------------------------------------------------------------------
extern "C" void kernel_launch(void* const* d_in, const int* in_sizes, int n_in,
                              void* d_out, int out_size, void* d_ws, size_t ws_size,
                              hipStream_t stream) {
  (void)in_sizes; (void)n_in; (void)out_size; (void)ws_size;
  const float* tf    = (const float*)d_in[1];  // (16,16,16,16,513)
  const float* imp   = (const float*)d_in[2];  // (16,16,16,16,16)
  const float* noise = (const float*)d_in[3];  // (16,16,16,16,1024)
  float* out = (float*)d_out;                  // (1,1,16384) f32

  char* ws = (char*)d_ws;
  float*  sf     = (float*)(ws + 0);          // 4096*1024*4  = 16 MB
  float*  sf_new = (float*)(ws + 16777216);   // 16 MB
  __bf16* sfh    = (__bf16*)(ws + 33554432);  // 8 MB
  __bf16* specRe = (__bf16*)(ws + 41943040);  // 4096*544*2
  __bf16* specIm = (__bf16*)(ws + 46399488);
  __bf16* bfc    = (__bf16*)(ws + 50855936);  // 544*1024*2 each
  __bf16* bfs    = (__bf16*)(ws + 51970048);
  __bf16* bic    = (__bf16*)(ws + 53084160);
  __bf16* bis    = (__bf16*)(ws + 54198272);

  basis_kernel<<<(NCOP * SPB + 255) / 256, 256, 0, stream>>>(bfc, bfs, bic, bis);
  for (int b = 0; b < NBLK; ++b) {
    add_env_kernel<<<NVOX * SPB / 256, 256, 0, stream>>>(imp, noise, sf, sfh, b, b == 0);
    fwd_dft_kernel<<<NVOX / 16, 256, 0, stream>>>(sfh, bfc, bfs, tf, specRe, specIm, b);
    inv_dft_kernel<<<NVOX / 16, 256, 0, stream>>>(specRe, specIm, bic, bis, sf_new);
    box_mic_kernel<<<NVOX * SPB / 256, 256, 0, stream>>>(sf_new, sf, out, b);
  }
}
